// GatedEquivariantBlock_52080773431358
// MI455X (gfx1250) — compile-verified, matrix operand order checked by reading, not run on hardware
//
#include <hip/hip_runtime.h>
#include <math.h>

typedef __attribute__((ext_vector_type(16))) _Float16 v16h;
typedef __attribute__((ext_vector_type(8)))  float    v8f;

#define NNODE  4096
#define NEDGE  32768
#define NBAS   8
#define HID    64
#define DIMIN  120
#define DIMSH  9
#define WN     3456
#define NINS   11
#define TMPTOT 592
#define CGTOT  363
#define NTILES (WN/16)          // 216 N-tiles of 16 weight columns

// Tensor-product instruction metadata (matches reference INSTR order).
// Plain constexpr: used for compile-time table generation + folded literals.
constexpr int k_wo[NINS]     = {0,1024,1536,1792,2048,2560,2688,2944,3008,3136,3392};
constexpr int k_m3[NINS]     = {32,16,8,16,32,8,16,8,16,32,8};
constexpr int k_d3[NINS]     = {1,3,5,3,1,5,3,5,3,1,5};
constexpr int k_toffs[NINS]  = {0,32,128,288,336,352,432,480,520,544,552};
constexpr int k_offout[NINS] = {0,32,80,32,0,80,32,80,32,0,80};

// Per-N-tile packed descriptor, generated at compile time:
// [10:0]=c0-wo  [12:11]=log2(m3)-3  [14:13]=d3code(d3=1+2*code)
// [24:15]=toff  [31:25]=offout
struct TileTab { unsigned d[NTILES]; };
constexpr TileTab mkTileTab() {
  TileTab t{};
  for (int nt = 0; nt < NTILES; ++nt) {
    int c0 = nt * 16;
    int ins = 0;
    while (ins + 1 < NINS && c0 >= k_wo[ins + 1]) ++ins;
    int scode  = (k_m3[ins] == 8) ? 0 : (k_m3[ins] == 16) ? 1 : 2;
    int d3code = (k_d3[ins] == 1) ? 0 : (k_d3[ins] == 3) ? 1 : 2;
    t.d[nt] = (unsigned)(c0 - k_wo[ins])
            | ((unsigned)scode  << 11)
            | ((unsigned)d3code << 13)
            | ((unsigned)k_toffs[ins]  << 15)
            | ((unsigned)k_offout[ins] << 25);
  }
  return t;
}
constexpr TileTab c_td = mkTileTab();

// Shared fragment index mapping (ISA 16-bit operand layout):
// element t of a v16h fragment, lane-half hi -> K index within a 32-wide slab.
__device__ __forceinline__ int fragK(int t, int hi) {
  return ((t >> 3) << 4) + (hi << 3) + (t & 7);
}

// ---------------------------------------------------------------------------
// Build real-basis Wigner-3j tables on device (l <= 2), unit Frobenius norm.
// l=1 basis order (y,z,x); l=2 basis order (xy, yz, 2z^2-x^2-y^2, zx, x^2-y^2).
// ---------------------------------------------------------------------------
__global__ void k_setup_cg(float* __restrict__ cg) {
  if (threadIdx.x != 0 || blockIdx.x != 0) return;
  const float is2 = 0.70710678f, is3 = 0.57735027f, is5 = 0.44721360f, is6 = 0.40824829f;
  float M[5][3][3] = {};                 // orthonormal symmetric-traceless basis (cart x,y,z)
  M[0][0][1] = M[0][1][0] = is2;         // xy
  M[1][1][2] = M[1][2][1] = is2;         // yz
  M[2][0][0] = -is6; M[2][1][1] = -is6; M[2][2][2] = 2.0f*is6;  // 2z^2-x^2-y^2
  M[3][2][0] = M[3][0][2] = is2;         // zx
  M[4][0][0] = is2;  M[4][1][1] = -is2;  // x^2-y^2
  const int p1[3] = {1, 2, 0};           // e3nn l=1 order (y,z,x) -> cart index

  cg[0] = 1.0f;                                                            // (0,0,0)
  for (int j = 0; j < 3; ++j) for (int k = 0; k < 3; ++k)
    cg[1 + j*3 + k] = (j == k) ? is3 : 0.0f;                               // (0,1,1)
  for (int j = 0; j < 5; ++j) for (int k = 0; k < 5; ++k)
    cg[10 + j*5 + k] = (j == k) ? is5 : 0.0f;                              // (0,2,2)
  for (int i = 0; i < 3; ++i) for (int k = 0; k < 3; ++k)
    cg[35 + i*3 + k] = (i == k) ? is3 : 0.0f;                              // (1,0,1)
  for (int i = 0; i < 3; ++i) for (int j = 0; j < 3; ++j)
    cg[44 + i*3 + j] = (i == j) ? is3 : 0.0f;                              // (1,1,0)
  for (int i = 0; i < 3; ++i) for (int j = 0; j < 3; ++j) for (int k = 0; k < 5; ++k)
    cg[53 + (i*3 + j)*5 + k] = M[k][p1[i]][p1[j]] * is5;                   // (1,1,2)
  for (int i = 0; i < 3; ++i) for (int j = 0; j < 5; ++j) for (int k = 0; k < 3; ++k)
    cg[98 + (i*5 + j)*3 + k] = M[j][p1[i]][p1[k]] * is5;                   // (1,2,1)
  for (int i = 0; i < 5; ++i) for (int k = 0; k < 5; ++k)
    cg[143 + i*5 + k] = (i == k) ? is5 : 0.0f;                             // (2,0,2)
  for (int i = 0; i < 5; ++i) for (int j = 0; j < 3; ++j) for (int k = 0; k < 3; ++k)
    cg[168 + (i*3 + j)*3 + k] = M[i][p1[j]][p1[k]] * is5;                  // (2,1,1)
  for (int i = 0; i < 5; ++i) for (int j = 0; j < 5; ++j)
    cg[213 + i*5 + j] = (i == j) ? is5 : 0.0f;                             // (2,2,0)
  // (2,2,2): fully symmetric invariant via symmetrized triple trace
  float T[5][5][5]; float nrm = 0.0f;
  for (int a = 0; a < 5; ++a) for (int b = 0; b < 5; ++b) for (int c = 0; c < 5; ++c) {
    float s = 0.0f;
    for (int p = 0; p < 3; ++p) for (int q = 0; q < 3; ++q) for (int r = 0; r < 3; ++r)
      s += M[a][p][q] * (M[b][q][r]*M[c][r][p] + M[c][q][r]*M[b][r][p]);
    T[a][b][c] = s; nrm += s*s;
  }
  float inv = 1.0f / sqrtf(nrm);
  for (int a = 0; a < 5; ++a) for (int b = 0; b < 5; ++b) for (int c = 0; c < 5; ++c)
    cg[238 + (a*5 + b)*5 + c] = T[a][b][c] * inv;
}

__global__ void k_zero(float* __restrict__ p, int n) {
  int i = blockIdx.x * blockDim.x + threadIdx.x;
  if (i < n) p[i] = 0.0f;
}

// Pre-swizzle w3 (64 x 3456 f32) into fp16 WMMA B-fragment layout:
// w3p[nt*1024 + lane*32 + frag*16 + t] = w3[(fragK(t,hi)+32*frag)*WN + nt*16 + n]
// so k_tp loads each lane's fragment as contiguous 32-byte vectors.
__global__ void k_w3pack(const float* __restrict__ w3, _Float16* __restrict__ w3p) {
  int idx = blockIdx.x * blockDim.x + threadIdx.x;
  if (idx >= NTILES * 1024) return;
  int t    = idx & 15;
  int frag = (idx >> 4) & 1;
  int lane = (idx >> 5) & 31;
  int nt   = idx >> 10;
  int hi = lane >> 4, n = lane & 15;
  int k = fragK(t, hi) + frag * 32;
  w3p[idx] = (_Float16)w3[k * WN + nt * 16 + n];
}

// Radial MLP layers 1-2 (tiny: ~0.3 GFLOP total) -> h in fp16, stored directly
// in WMMA A-fragment layout: hp[tile*1024 + lane*32 + frag*16 + t].
__global__ void k_mlp12(const float* __restrict__ rad,
                        const float* __restrict__ w1, const float* __restrict__ b1,
                        const float* __restrict__ w2, const float* __restrict__ b2,
                        _Float16* __restrict__ hp) {
  __shared__ float sw1[NBAS*HID], sb1[HID], sw2[HID*HID], sb2[HID];
  for (int i = threadIdx.x; i < NBAS*HID; i += blockDim.x) sw1[i] = w1[i];
  for (int i = threadIdx.x; i < HID*HID;  i += blockDim.x) sw2[i] = w2[i];
  for (int i = threadIdx.x; i < HID;      i += blockDim.x) { sb1[i] = b1[i]; sb2[i] = b2[i]; }
  __syncthreads();
  int e = blockIdx.x * blockDim.x + threadIdx.x;
  if (e >= NEDGE) return;
  float r[NBAS];
  for (int i = 0; i < NBAS; ++i) r[i] = rad[e*NBAS + i];
  float h1[HID];
  for (int j = 0; j < HID; ++j) {
    float s = sb1[j];
    for (int i = 0; i < NBAS; ++i) s += r[i] * sw1[i*HID + j];
    h1[j] = s / (1.0f + expf(-s));                       // silu
  }
  const int tile = e >> 4, m = e & 15;
  for (int j = 0; j < HID; ++j) {
    float s = sb2[j];
    for (int u = 0; u < HID; ++u) s += h1[u] * sw2[u*HID + j];
    s = s / (1.0f + expf(-s));
    // scatter into A-fragment layout (inverse of fragK)
    int frag = j >> 5, kl = j & 31;
    int hi = (kl >> 3) & 1;
    int t  = ((kl >> 4) << 3) + (kl & 7);
    hp[tile*1024 + (m + (hi << 4))*32 + frag*16 + t] = (_Float16)s;
  }
}

// Per-instruction tmp stage, all shapes compile-time:
// tmp[e,u,k] = sum_{i,j} x1[offin + u*D1 + i] * x2[offsh + j] * cg[(i*D2+j)*D3 + k]
// lane&15 = edge, the two lane-halves split the M1*D3 index range (no division).
template<int D1, int D2, int D3, int M1>
__device__ __forceinline__ void tmp_stage(int lane, int toff, int cgoff, int offin, int offsh,
                                          const float* __restrict__ cgs,
                                          const float* __restrict__ x1s,
                                          const float* __restrict__ x2s,
                                          float* __restrict__ tmps) {
  const int el = lane & 15;
  const float* xx0 = &x1s[el*DIMIN + offin];
  const float* yy  = &x2s[el*DIMSH + offsh];
  const float* cg  = &cgs[cgoff];
  for (int idx = lane >> 4; idx < M1*D3; idx += 2) {
    int u = idx / D3, k = idx % D3;      // compile-time divisor
    float s = 0.0f;
    #pragma unroll
    for (int i = 0; i < D1; ++i) {
      float xi = xx0[u*D1 + i];
      #pragma unroll
      for (int j = 0; j < D2; ++j) s += xi * yy[j] * cg[(i*D2 + j)*D3 + k];
    }
    tmps[el*TMPTOT + toff + idx] = s;
  }
}

// Fully-unrolled TP consumer for one N-tile: hoist all tmp reads (vectorizable
// contiguous LDS loads, one wait) then fire the ds_add_f32 atomics.
template<int D3>
__device__ __forceinline__ void tp_consume(const v8f& acc, int hi, float coeff,
                                           int tbase, int obase,
                                           const float* __restrict__ tmps,
                                           float* __restrict__ msgs) {
  float tv[8][D3];
  #pragma unroll
  for (int r = 0; r < 8; ++r) {
    int el = r + (hi << 3);
    #pragma unroll
    for (int k = 0; k < D3; ++k) tv[r][k] = tmps[el*TMPTOT + tbase + k];
  }
  #pragma unroll
  for (int r = 0; r < 8; ++r) {
    int el = r + (hi << 3);
    float wt = acc[r] * coeff;
    #pragma unroll
    for (int k = 0; k < D3; ++k)
      atomicAdd(&msgs[el*DIMIN + obase + k], wt * tv[r][k]);
  }
}

// ---------------------------------------------------------------------------
// Fused: weight GEMM (h @ w3 + b3 via WMMA f16, M=16 edges, K=64, N=3456)
//        -> CG tensor product -> segment-sum scatter. One wave per 16 edges;
//        weight tiles are consumed straight out of the WMMA accumulator.
// ---------------------------------------------------------------------------
__global__ void __launch_bounds__(32) k_tp(const float* __restrict__ nf,
                                           const int* __restrict__ eidx,
                                           const float* __restrict__ sh,
                                           const _Float16* __restrict__ hp,
                                           const _Float16* __restrict__ w3p,
                                           const float* __restrict__ b3,
                                           const float* __restrict__ cgbuf,
                                           float* __restrict__ agg) {
  __shared__ float x1s[16*DIMIN];     // gathered node features (16 edges)
  __shared__ float x2s[16*DIMSH];     // edge SH
  __shared__ float tmps[16*TMPTOT];   // tmp[e][u,k] per instruction
  __shared__ float msgs[16*DIMIN];    // per-edge output messages
  __shared__ float cgs[CGTOT];
  const int lane = threadIdx.x;
  const int e0   = blockIdx.x * 16;
  const int* src = eidx;              // edge_index[0]
  const int* dst = eidx + NEDGE;      // edge_index[1]

  for (int i = lane; i < CGTOT; i += 32) cgs[i] = cgbuf[i];
  for (int i = lane; i < 16*DIMSH; i += 32) x2s[i] = sh[e0*DIMSH + i];
  for (int v = lane; v < 16*DIMIN; v += 32) {
    int el = v / DIMIN, j = v % DIMIN;
    x1s[v]  = nf[src[e0 + el]*DIMIN + j];
    msgs[v] = 0.0f;
  }
  __syncthreads();

  // Per-instruction tmp stages (all shapes static).
  tmp_stage<1,1,1,32>(lane,   0,   0,  0, 0, cgs, x1s, x2s, tmps);
  tmp_stage<1,3,3,32>(lane,  32,   1,  0, 1, cgs, x1s, x2s, tmps);
  tmp_stage<1,5,5,32>(lane, 128,  10,  0, 4, cgs, x1s, x2s, tmps);
  tmp_stage<3,1,3,16>(lane, 288,  35, 32, 0, cgs, x1s, x2s, tmps);
  tmp_stage<3,3,1,16>(lane, 336,  44, 32, 1, cgs, x1s, x2s, tmps);
  tmp_stage<3,3,5,16>(lane, 352,  53, 32, 1, cgs, x1s, x2s, tmps);
  tmp_stage<3,5,3,16>(lane, 432,  98, 32, 4, cgs, x1s, x2s, tmps);
  tmp_stage<5,1,5, 8>(lane, 480, 143, 80, 0, cgs, x1s, x2s, tmps);
  tmp_stage<5,3,3, 8>(lane, 520, 168, 80, 1, cgs, x1s, x2s, tmps);
  tmp_stage<5,5,1, 8>(lane, 544, 213, 80, 4, cgs, x1s, x2s, tmps);
  tmp_stage<5,5,5, 8>(lane, 552, 238, 80, 4, cgs, x1s, x2s, tmps);
  __syncthreads();

  // A fragments: packed by k_mlp12, two contiguous 32B vector loads per lane.
  const int hi = lane >> 4, nn = lane & 15;
  const v16h* af = (const v16h*)hp + (size_t)blockIdx.x*64 + lane*2;
  v16h aLo = af[0];
  v16h aHi = af[1];
  const v16h* bf0 = (const v16h*)w3p;

  // 216 N-tiles of 16 weight columns; wave-uniform instruction per tile.
  #pragma unroll 2
  for (int nt = 0; nt < NTILES; ++nt) {
    // Compile-time packed descriptor: one scalar load, shift/mask decode.
    const unsigned td = c_td.d[nt];
    const int locbase = td & 0x7FF;
    const int shift   = 3 + ((td >> 11) & 3);
    const int d3code  = (td >> 13) & 3;
    const int toff    = (td >> 15) & 0x3FF;
    const int offout  = td >> 25;
    const int d3      = 1 + 2*d3code;
    const float coeff = (d3code == 0) ? 0.13363062f
                      : (d3code == 1) ? 0.20412415f : 0.27950850f;
    const int loc   = locbase + nn;
    const int u     = loc >> shift;            // m3 is a power of two
    const int w     = loc & ((1 << shift) - 1);
    const int tbase = toff + u*d3;
    const int obase = offout + w*d3;

    // B fragments: pre-swizzled, two contiguous 32B vector loads per lane.
    const v16h* bf = bf0 + nt*64 + lane*2;
    v16h bLo = bf[0];
    v16h bHi = bf[1];
    if (nt + 4 < NTILES)
      __builtin_prefetch((const void*)(bf0 + (nt + 4)*64 + lane*2), 0, 1);

    float bias = b3[nt*16 + nn];
    v8f acc;
    #pragma unroll
    for (int r = 0; r < 8; ++r) acc[r] = bias;   // C init = b3 (same col all rows)
    acc = __builtin_amdgcn_wmma_f32_16x16x32_f16(false, aLo, false, bLo,
                                                 (short)0, acc, false, false);
    acc = __builtin_amdgcn_wmma_f32_16x16x32_f16(false, aHi, false, bHi,
                                                 (short)0, acc, false, false);

    // Consume accumulator in place: C layout lane -> (M = r + 8*hi, N = lane&15).
    switch (d3code) {
      case 0:  tp_consume<1>(acc, hi, coeff, tbase, obase, tmps, msgs); break;
      case 1:  tp_consume<3>(acc, hi, coeff, tbase, obase, tmps, msgs); break;
      default: tp_consume<5>(acc, hi, coeff, tbase, obase, tmps, msgs); break;
    }
  }
  __syncthreads();

  // segment_sum scatter into agg
  for (int v = lane; v < 16*DIMIN; v += 32) {
    int el = v / DIMIN, j = v % DIMIN;
    atomicAdd(&agg[dst[e0 + el]*DIMIN + j], msgs[v]);
  }
}

// Self-interaction (o3.Linear per l, scaled 1/sqrt(m)) + residual.
__global__ void k_si(const float* __restrict__ agg, const float* __restrict__ nf,
                     const float* __restrict__ w0, const float* __restrict__ w1,
                     const float* __restrict__ w2, float* __restrict__ out) {
  __shared__ float s0[32*32], s1[16*16], s2[8*8];
  for (int i = threadIdx.x; i < 32*32; i += blockDim.x) s0[i] = w0[i];
  for (int i = threadIdx.x; i < 16*16; i += blockDim.x) s1[i] = w1[i];
  for (int i = threadIdx.x; i < 8*8;   i += blockDim.x) s2[i] = w2[i];
  __syncthreads();
  int n = blockIdx.x * blockDim.x + threadIdx.x;
  if (n >= NNODE) return;
  const float* a = &agg[n*DIMIN];
  const float* x = &nf[n*DIMIN];
  float* o = &out[n*DIMIN];
  const float i32s = 0.17677670f, i16s = 0.25f, i8s = 0.35355339f;
  for (int v = 0; v < 32; ++v) {
    float s = 0.0f;
    for (int u = 0; u < 32; ++u) s += a[u] * s0[u*32 + v];
    o[v] = s * i32s + x[v];
  }
  for (int v = 0; v < 16; ++v)
    for (int k = 0; k < 3; ++k) {
      float s = 0.0f;
      for (int u = 0; u < 16; ++u) s += a[32 + u*3 + k] * s1[u*16 + v];
      o[32 + v*3 + k] = s * i16s + x[32 + v*3 + k];
    }
  for (int v = 0; v < 8; ++v)
    for (int k = 0; k < 5; ++k) {
      float s = 0.0f;
      for (int u = 0; u < 8; ++u) s += a[80 + u*5 + k] * s2[u*8 + v];
      o[80 + v*5 + k] = s * i8s + x[80 + v*5 + k];
    }
}

extern "C" void kernel_launch(void* const* d_in, const int* in_sizes, int n_in,
                              void* d_out, int out_size, void* d_ws, size_t ws_size,
                              hipStream_t stream) {
  const float* nf   = (const float*)d_in[0];
  const int*   eidx = (const int*)  d_in[1];
  const float* sh   = (const float*)d_in[2];
  const float* rad  = (const float*)d_in[3];
  const float* w1   = (const float*)d_in[4];
  const float* b1   = (const float*)d_in[5];
  const float* w2   = (const float*)d_in[6];
  const float* b2   = (const float*)d_in[7];
  const float* w3   = (const float*)d_in[8];
  const float* b3   = (const float*)d_in[9];
  const float* siw0 = (const float*)d_in[10];
  const float* siw1 = (const float*)d_in[11];
  const float* siw2 = (const float*)d_in[12];
  float* out = (float*)d_out;

  // Workspace layout (all 256B aligned):
  char* ws = (char*)d_ws;
  _Float16* hp   = (_Float16*)(ws);                       // E*64 fp16 packed A frags = 4 MB
  _Float16* w3p  = (_Float16*)(ws + 4194304);             // 216*1024 fp16 packed B frags = 432 KB
  float*    agg  = (float*)   (ws + 4194304 + 442368);    // N*120 f32 ~ 1.9 MB
  float*    cg   = (float*)   (ws + 6602752);             // 363 f32

  k_setup_cg<<<1, 1, 0, stream>>>(cg);
  k_zero<<<(NNODE*DIMIN + 255)/256, 256, 0, stream>>>(agg, NNODE*DIMIN);
  k_w3pack<<<(NTILES*1024 + 255)/256, 256, 0, stream>>>(w3, w3p);
  k_mlp12<<<NEDGE/128, 128, 0, stream>>>(rad, w1, b1, w2, b2, hp);
  k_tp<<<NEDGE/16, 32, 0, stream>>>(nf, eidx, sh, hp, w3p, b3, cg, agg);
  k_si<<<(NNODE + 127)/128, 128, 0, stream>>>(agg, nf, siw0, siw1, siw2, out);
}
